// Alignment_vector_19481971655017
// MI455X (gfx1250) — compile-verified, matrix operand order checked by reading, not run on hardware
//
#include <hip/hip_runtime.h>
#include <math.h>

#ifndef __has_builtin
#define __has_builtin(x) 0
#endif

// ---------------- CDNA5 WMMA / TDM types ----------------
typedef __attribute__((ext_vector_type(16))) __bf16    v16bf;
typedef __attribute__((ext_vector_type(8)))  float     v8f;
typedef __attribute__((ext_vector_type(4)))  unsigned  v4u;
typedef __attribute__((ext_vector_type(8)))  int       v8i;
typedef __attribute__((ext_vector_type(4)))  int       v4i;

// Pick the best available LDS-fill path for the context slab.
#if __has_builtin(__builtin_amdgcn_tensor_load_to_lds) && __has_builtin(__builtin_amdgcn_s_wait_tensorcnt)
#define CTX_PATH_TDM 1
#elif __has_builtin(__builtin_amdgcn_global_load_async_to_lds_b128)
#define CTX_PATH_ASYNC 1
#endif

// Problem constants (from reference setup_inputs)
constexpr int B_  = 128;   // batch
constexpr int LQ  = 128;   // query length
constexpr int LS  = 256;   // context length
constexpr int D   = 1024;  // feature dim
constexpr int SD  = 256;   // sim_dim
constexpr int NT  = 512;   // threads per block (16 waves of 32)

// LDS layout (bytes)
constexpr size_t OFF_ATTN  = 0;                  // 128*256*2  = 65536  (bf16 attn)
constexpr size_t OFF_WC    = OFF_ATTN + 65536;   // 64*1024*2  = 131072 (bf16 wcontext half)
constexpr size_t OFF_STAGE = OFF_WC + 131072;    // 73728: 2x 32KB fp32 ctx slabs + 8KB q slab
constexpr size_t OFF_NORMS = OFF_STAGE + 73728;  // 256*4
constexpr size_t OFF_RED   = OFF_NORMS + 1024;   // 64*4
constexpr size_t SMEM_BYTES = OFF_RED + 256;     // 271616 B (<= 320KB/WGP)

__device__ __forceinline__ unsigned pack2bf(float a, float b) {
  union { __bf16 h[2]; unsigned u; } p;
  p.h[0] = (__bf16)a; p.h[1] = (__bf16)b;
  return p.u;
}

// ISA 7.12.2: 16-bit A/B 16x32 fragment. lane&15 = matrix row (A: M, B: N via B^T),
// VGPR v holds K pair {2v,2v+1} (v<4) or {16+2(v-4),..}; +8 for lanes 16..31.
__device__ __forceinline__ int frag_kb(int v, int half) {
  return (v < 4 ? 2 * v : 16 + 2 * (v - 4)) + 8 * half;
}

__device__ __forceinline__ v16bf frag_bf16(const __bf16* p, int row, int stride, int half) {
  union { v16bf v; unsigned u[8]; } f;
  const __bf16* rp = p + row * stride;
#pragma unroll
  for (int r = 0; r < 8; ++r) f.u[r] = *(const unsigned*)(rp + frag_kb(r, half));
  return f.v;
}

__device__ __forceinline__ v16bf frag_f32(const float* p, int row, int stride, int half) {
  union { v16bf v; unsigned u[8]; } f;
  const float* rp = p + row * stride;
#pragma unroll
  for (int r = 0; r < 8; ++r) {
    int kb = frag_kb(r, half);
    f.u[r] = pack2bf(rp[kb], rp[kb + 1]);
  }
  return f.v;
}

__device__ __forceinline__ v8f wmma_bf16(v16bf a, v16bf b, v8f c) {
  return __builtin_amdgcn_wmma_f32_16x16x32_bf16(false, a, false, b, (short)0, c, false, false);
}

// ---- Context-slab staging: [256 rows][32 fp32 cols], row stride D in global ----
__device__ __forceinline__ void ctx_stage_issue(float* dst, const float* gsrc, int tid) {
#if defined(CTX_PATH_TDM)
  if (tid == 0) {
    // Tensor DMA descriptor (ISA 8.3/8.4): 2D tile 32 x 256 of 4-byte elements.
    unsigned lofs = (unsigned)(unsigned long long)(__attribute__((address_space(3))) float*)dst;
    unsigned long long ga = (unsigned long long)gsrc;
    v4u g0;
    g0.x = 1u;                                   // count=1, user mode, no gather
    g0.y = lofs;                                 // lds_addr (bytes)
    g0.z = (unsigned)ga;                         // global_addr[31:0]
    g0.w = (unsigned)((ga >> 32) & 0x01FFFFFFull) | (2u << 30);  // addr[56:32] | type=2
    v8i g1;
    g1[0] = (int)(2u << 16);        // wg_mask=0 | data_size=2 (4 bytes)
    g1[1] = (int)(32u << 16);       // tensor_dim0[15:0]=32 -> bits[63:48]
    g1[2] = (int)(256u << 16);      // tensor_dim0 hi=0 | tensor_dim1[15:0]=256
    g1[3] = (int)(32u << 16);       // tensor_dim1 hi=0 | tile_dim0=32
    g1[4] = (int)256;               // tile_dim1=256 | tile_dim2=0
    g1[5] = (int)1024;              // tensor_dim0_stride[31:0] = 1024 elements
    g1[6] = 0;                      // stride0 hi | stride1 lo
    g1[7] = 0;
    v4i z = {0, 0, 0, 0};
#if __clang_major__ >= 23
    v8i z8 = {0, 0, 0, 0, 0, 0, 0, 0};
    __builtin_amdgcn_tensor_load_to_lds(g0, g1, z, z, z8, 0);
#else
    __builtin_amdgcn_tensor_load_to_lds(g0, g1, z, z, 0);
#endif
  }
#elif defined(CTX_PATH_ASYNC)
  for (int idx = tid; idx < 2048; idx += NT) {   // 4 x b128 per thread
    int row = idx >> 3, c4 = (idx & 7) << 2;
    const float* g = gsrc + (size_t)row * D + c4;
    float* l = dst + row * 32 + c4;
    __builtin_amdgcn_global_load_async_to_lds_b128(
        (const __attribute__((address_space(1))) void*)g,
        (__attribute__((address_space(3))) void*)l, 0, 0);
  }
#else
  for (int idx = tid; idx < 2048; idx += NT) {
    int row = idx >> 3, c4 = (idx & 7) << 2;
    *(float4*)(dst + row * 32 + c4) = *(const float4*)(gsrc + (size_t)row * D + c4);
  }
#endif
}

__device__ __forceinline__ void ctx_stage_wait(bool more, int tid) {
#if defined(CTX_PATH_TDM)
  if (tid == 0) {
    if (more) __builtin_amdgcn_s_wait_tensorcnt(1);   // allow next slab in flight
    else      __builtin_amdgcn_s_wait_tensorcnt(0);
  }
#elif defined(CTX_PATH_ASYNC)
#if __has_builtin(__builtin_amdgcn_s_wait_asynccnt)
  if (more) __builtin_amdgcn_s_wait_asynccnt(4);      // 4 b128 ops per wave per slab
  else      __builtin_amdgcn_s_wait_asynccnt(0);
#else
  if (more) asm volatile("s_wait_asynccnt 0x4" ::: "memory");
  else      asm volatile("s_wait_asynccnt 0x0" ::: "memory");
#endif
#endif
  (void)more; (void)tid;
}

__global__ __launch_bounds__(NT, 1)
void align_fused_kernel(const float* __restrict__ query,
                        const float* __restrict__ context,
                        const float* __restrict__ matrix,
                        const float* __restrict__ Wm,
                        const float* __restrict__ bias,
                        const int*   __restrict__ smoothp,
                        float* __restrict__ out) {
  extern __shared__ char smem[];
  __bf16* attnb = (__bf16*)(smem + OFF_ATTN);   // [LQ][LS] bf16
  __bf16* wc    = (__bf16*)(smem + OFF_WC);     // [64][D] bf16 (one Lq half)
  char*   stage = smem + OFF_STAGE;
  float*  norms = (float*)(smem + OFF_NORMS);   // [LS]
  float*  red   = (float*)(smem + OFF_RED);     // [64]

  const int tid  = threadIdx.x;
  const int lane = tid & 31;
  const int w    = tid >> 5;
  const int half = lane >> 4;
  const int mrow = lane & 15;
  const int b    = blockIdx.x;
  const float smooth = (float)(*smoothp);

  // ============ Phase 1: S = context * (query*matrix)^T, LeakyReLU, store transposed ============
  {
    float*  sCtx0 = (float*)stage;                 // [256][32] fp32, buffer 0
    float*  sCtx1 = (float*)(stage + 32768);       // buffer 1
    __bf16* sQ    = (__bf16*)(stage + 65536);      // [128][32] bf16
    const float* gctx = context + (size_t)b * LS * D;

    v8f acc[8];
#pragma unroll
    for (int i = 0; i < 8; ++i) acc[i] = {};

    ctx_stage_issue(sCtx0, gctx, tid);             // preload slab 0 (DMA overlaps sQ staging)
#pragma unroll 1
    for (int ks = 0; ks < D; ks += 32) {
      const int it = ks >> 5;
      float* cur = (it & 1) ? sCtx1 : sCtx0;
      float* nxt = (it & 1) ? sCtx0 : sCtx1;
      __syncthreads();                             // prev compute done: safe to restage sQ / land nxt
      for (int i = tid; i < LQ * 32; i += NT) {
        int r = i >> 5, c = i & 31;
        size_t gi = ((size_t)b * LQ + r) * D + ks + c;
        sQ[i] = (__bf16)(query[gi] * matrix[gi]);
      }
      const bool more = (ks + 32) < D;
      if (more) ctx_stage_issue(nxt, gctx + ks + 32, tid);  // prefetch next slab via DMA
      ctx_stage_wait(more, tid);                   // wait current slab only
      __syncthreads();
      v16bf afrag = frag_f32(cur, 16 * w + mrow, 32, half); // wave w owns s-tile w
#pragma unroll
      for (int tn = 0; tn < 8; ++tn) {
        v16bf bfrag = frag_bf16(sQ, tn * 16 + mrow, 32, half);
        acc[tn] = wmma_bf16(afrag, bfrag, acc[tn]);
      }
    }
    // LeakyReLU(0.1) + transposed store attn[q][s] (bf16)
#pragma unroll
    for (int tn = 0; tn < 8; ++tn)
#pragma unroll
      for (int r = 0; r < 8; ++r) {
        int s = 16 * w + r + 8 * half;
        int q = tn * 16 + mrow;
        float v = acc[tn][r];
        v = (v >= 0.f) ? v : 0.1f * v;
        attnb[q * LS + s] = (__bf16)v;
      }
    __syncthreads();
  }

  // ============ Phase 2: l2norm over Lq (per s), softmax*smooth over Ls (per q) ============
  if (tid < LS) {
    float ss = 0.f;
    for (int q = 0; q < LQ; ++q) { float v = (float)attnb[q * LS + tid]; ss += v * v; }
    norms[tid] = sqrtf(ss) + 1e-8f;
  }
  __syncthreads();
#pragma unroll 1
  for (int i = 0; i < 8; ++i) {
    int qrow = w * 8 + i;
    float v[8]; float mx = -3.0e38f;
#pragma unroll
    for (int j = 0; j < 8; ++j) {
      int s = lane + 32 * j;
      v[j] = (float)attnb[qrow * LS + s] / norms[s] * smooth;
      mx = fmaxf(mx, v[j]);
    }
#pragma unroll
    for (int d = 16; d > 0; d >>= 1) mx = fmaxf(mx, __shfl_xor(mx, d, 32));
    float sum = 0.f;
#pragma unroll
    for (int j = 0; j < 8; ++j) { v[j] = __expf(v[j] - mx); sum += v[j]; }
#pragma unroll
    for (int d = 16; d > 0; d >>= 1) sum += __shfl_xor(sum, d, 32);
    float inv = 1.f / sum;
#pragma unroll
    for (int j = 0; j < 8; ++j) attnb[qrow * LS + lane + 32 * j] = (__bf16)(v[j] * inv);
  }
  __syncthreads();

  // ============ Phases 3+4 per Lq half (wcontext half fits LDS as bf16 64x1024) ============
#pragma unroll 1
  for (int qh = 0; qh < 2; ++qh) {
    // ---- Phase 3: wcontext = attn * context ; accumulate row sumsq ----
    __bf16* ctxT = (__bf16*)stage;                 // [64][256] context^T tile (n-major)
    if (tid < 64) red[tid] = 0.f;
    __syncthreads();
    const int tm3 = w >> 2, tn3 = w & 3;
#pragma unroll 1
    for (int n0 = 0; n0 < D; n0 += 64) {
      for (int i = tid; i < 64 * LS; i += NT) {
        int s = i >> 6, nl = i & 63;               // coalesced read, transposed LDS write
        ctxT[nl * LS + s] = (__bf16)context[((size_t)b * LS + s) * D + n0 + nl];
      }
      __syncthreads();
      v8f c = {};
#pragma unroll 1
      for (int k0 = 0; k0 < LS; k0 += 32) {
        v16bf a  = frag_bf16(attnb + k0, qh * 64 + tm3 * 16 + mrow, LS, half);
        v16bf bb = frag_bf16(ctxT + k0, tn3 * 16 + mrow, LS, half);
        c = wmma_bf16(a, bb, c);
      }
#pragma unroll
      for (int r = 0; r < 8; ++r) {
        int ql = tm3 * 16 + r + 8 * half;
        int n  = tn3 * 16 + mrow;
        float v = c[r];
        wc[ql * D + n0 + n] = (__bf16)v;
        atomicAdd(&red[ql], v * v);                // ds_add_f32
      }
      __syncthreads();
    }
    if (tid < 64) red[tid] = 1.f / (sqrtf(red[tid]) + 1e-8f);
    __syncthreads();

    // ---- Phase 4: sim = (query - wc/||wc||)^2 * W^T + b ; l2norm ; writeout ----
    __bf16* sD2 = (__bf16*)stage;                  // [64][64]
    __bf16* sW  = (__bf16*)(stage + 8192);         // [256][64]
    v8f s4[4];
#pragma unroll
    for (int i = 0; i < 4; ++i) s4[i] = {};
#pragma unroll 1
    for (int k0 = 0; k0 < D; k0 += 64) {
      for (int i = tid; i < 64 * 64; i += NT) {
        int ql = i >> 6, dl = i & 63;
        float wcv  = (float)wc[ql * D + k0 + dl] * red[ql];
        float qv   = query[((size_t)b * LQ + qh * 64 + ql) * D + k0 + dl];
        float diff = qv - wcv;
        sD2[i] = (__bf16)(diff * diff);
      }
      for (int i = tid; i < SD * 64; i += NT) {
        int kk = i >> 6, dl = i & 63;
        sW[i] = (__bf16)Wm[(size_t)kk * D + k0 + dl];
        if (k0 + 64 < D) __builtin_prefetch(&Wm[(size_t)kk * D + k0 + 64 + dl], 0, 0);
      }
      __syncthreads();
#pragma unroll
      for (int kk0 = 0; kk0 < 64; kk0 += 32) {
        v16bf bb = frag_bf16(sW + kk0, w * 16 + mrow, 64, half);   // wave w owns kk-tile w
#pragma unroll
        for (int tm = 0; tm < 4; ++tm) {
          v16bf a = frag_bf16(sD2 + kk0, tm * 16 + mrow, 64, half);
          s4[tm] = wmma_bf16(a, bb, s4[tm]);
        }
      }
      __syncthreads();
    }
    if (tid < 64) red[tid] = 0.f;
    __syncthreads();
#pragma unroll
    for (int tm = 0; tm < 4; ++tm)
#pragma unroll
      for (int r = 0; r < 8; ++r) {
        int ql = tm * 16 + r + 8 * half;
        int kk = w * 16 + mrow;
        float v = s4[tm][r] + bias[kk];
        atomicAdd(&red[ql], v * v);
      }
    __syncthreads();
    if (tid < 64) red[tid] = 1.f / (sqrtf(red[tid]) + 1e-8f);
    __syncthreads();
#pragma unroll
    for (int tm = 0; tm < 4; ++tm)
#pragma unroll
      for (int r = 0; r < 8; ++r) {
        int ql = tm * 16 + r + 8 * half;
        int kk = w * 16 + mrow;
        float v = s4[tm][r] + bias[kk];
        out[((size_t)b * LQ + qh * 64 + ql) * SD + kk] = v * red[ql];
      }
    __syncthreads();                               // next half reuses red/stage/wc
  }
}

extern "C" void kernel_launch(void* const* d_in, const int* in_sizes, int n_in,
                              void* d_out, int out_size, void* d_ws, size_t ws_size,
                              hipStream_t stream) {
  (void)in_sizes; (void)n_in; (void)out_size; (void)d_ws; (void)ws_size;
  const float* query   = (const float*)d_in[0];
  const float* context = (const float*)d_in[1];
  const float* matrix  = (const float*)d_in[2];
  const float* Wm      = (const float*)d_in[3];
  const float* bias    = (const float*)d_in[4];
  const int*   smooth  = (const int*)d_in[5];
  float* out = (float*)d_out;

  hipFuncSetAttribute(reinterpret_cast<const void*>(align_fused_kernel),
                      hipFuncAttributeMaxDynamicSharedMemorySize, (int)SMEM_BYTES);

  align_fused_kernel<<<dim3(B_), dim3(NT), SMEM_BYTES, stream>>>(
      query, context, matrix, Wm, bias, smooth, out);
}
